// ObjectDetector_44822278701603
// MI455X (gfx1250) — matrix operand order repaired and checked by reading, not compiled, not run on hardware
//
#include <hip/hip_runtime.h>
#include <cstdint>
#include <cstddef>

// ---------------------------------------------------------------- types
typedef __attribute__((ext_vector_type(16))) _Float16 v16h;
typedef __attribute__((ext_vector_type(8)))  _Float16 v8h;
typedef __attribute__((ext_vector_type(8)))  float    v8f;

// ---------------------------------------------------------------- consts
constexpr int FH = 62, FW = 62;           // rpn feature map (VALID 3x3 on 64x64)
constexpr int NPOS = FH * FW;             // 3844
constexpr int NANCH = NPOS * 9;           // 34596
constexpr size_t IMG_N  = size_t(1024) * 1024 * 3;
constexpr size_t W1_N   = 768 * 512;
constexpr size_t FEAT_N = 4096 * 512;
constexpr size_t W2_N   = 4608 * 512;
constexpr size_t RPN_N  = size_t(NPOS) * 512;

constexpr size_t ALGN(size_t x) { return (x + 255) & ~size_t(255); }
constexpr size_t O_IMGH = 0;
constexpr size_t O_W1P  = ALGN(O_IMGH + IMG_N * 2);   // packed fragment-order weights
constexpr size_t O_FEATH= ALGN(O_W1P  + W1_N  * 2);
constexpr size_t O_W2P  = ALGN(O_FEATH+ FEAT_N* 2);
constexpr size_t O_RPNH = ALGN(O_W2P  + W2_N  * 2);
constexpr size_t O_PROB = ALGN(O_RPNH + RPN_N * 2);
constexpr size_t O_DELT = ALGN(O_PROB + size_t(NANCH) * 4);
constexpr size_t O_BOX  = ALGN(O_DELT + size_t(NANCH) * 16);
constexpr size_t O_SC   = ALGN(O_BOX  + size_t(NANCH) * 16);
constexpr size_t O_CMAP = ALGN(O_SC   + size_t(NANCH) * 4);
constexpr size_t O_RMAP = ALGN(O_CMAP + size_t(4096) * 27 * 4);
constexpr size_t O_CSC  = ALGN(O_RMAP + size_t(4096) * 36 * 4);
constexpr size_t O_BBP  = ALGN(O_CSC  + 100 * 3 * 4);
constexpr size_t O_FBX  = ALGN(O_BBP  + 100 * 4 * 4);
constexpr size_t O_FSC  = ALGN(O_FBX  + 100 * 4 * 4);
constexpr size_t O_FLB  = ALGN(O_FSC  + 100 * 4);

// ---------------------------------------------------------------- f32 -> f16
__global__ void cvt_f16_kernel(const float* __restrict__ in, _Float16* __restrict__ out, int n) {
    int i = blockIdx.x * blockDim.x + threadIdx.x;
    if (i < n) out[i] = (_Float16)in[i];
}

// ---------------------------------------------------------------- weight pack: f32 [K,512] row-major
// -> f16 fragment order: [kstep][nsub(32)][lane(32)][e(16)], so a wave's B fragment is 32
// contiguous 32B lane-records (coalesced global b128 loads, no LDS transpose needed).
// B fragment layout: lanes 0-15 hold N=lane, K in {0..7,16..23}; lanes 16-31 N=lane-16, K in {8..15,24..31}.
__global__ void pack_w_kernel(const float* __restrict__ w, _Float16* __restrict__ out, int total) {
    int idx = blockIdx.x * blockDim.x + threadIdx.x;
    if (idx >= total) return;
    int k = idx >> 9, n = idx & 511;
    int kstep = k >> 5, kk = k & 31;
    int nsub = n >> 4, nn = n & 15;
    int lane = nn + (((kk >> 3) & 1) << 4);
    int e = (kk & 7) + ((kk >> 4) << 3);
    out[(((size_t)kstep * 32 + nsub) * 32 + lane) * 16 + e] = (_Float16)w[idx];
}

// ---------------------------------------------------------------- WMMA fragment-layout GEMMs
// A fragment (16x32 f16): lanes 0-15 hold M=lane, K in {0..7,16..23}; lanes 16-31 M=lane-16,
// K in {8..15,24..31}; element order within v16h: e0..7 = first K-octet, e8..15 = second.
// A tiles are staged in LDS already per-lane-contiguous so fragment load = 2x ds_load_b128.
constexpr int FRAG_STRIDE = 24;                 // halves per lane slot (48B, keeps 16B align)
constexpr int AS_HALVES = 4 * 32 * FRAG_STRIDE; // 4 m-subtiles

__device__ inline v16h load_frag(const _Float16* p) {
    v8h lo = *(const v8h*)p;
    v8h hi = *(const v8h*)(p + 8);
    return __builtin_shufflevector(lo, hi, 0, 1, 2, 3, 4, 5, 6, 7, 8, 9, 10, 11, 12, 13, 14, 15);
}

// map (row-in-tile 0..63, kk 0..31) -> fragment-layout LDS index
__device__ inline int a_slot(int r, int kk) {
    int msub = r >> 4, mm = r & 15, g = kk >> 3;
    return ((msub * 32) + mm + (g & 1) * 16) * FRAG_STRIDE + (kk & 7) + (g >> 1) * 8;
}

// backbone conv as GEMM: M=4096 (64x64 out pixels), K=768 (16x16x3 patch), N=512
__global__ __launch_bounds__(256) void gemm_backbone(
    const _Float16* __restrict__ img, const _Float16* __restrict__ bpack,
    const float* __restrict__ bias, _Float16* __restrict__ out)
{
    __shared__ _Float16 As_f[AS_HALVES];
    const int tid = threadIdx.x;
    const int bm = blockIdx.x, bn = blockIdx.y;
    const int wave = tid >> 5, lane = tid & 31;
    const int wm = wave & 1, wn = wave >> 1;

    v8f acc[2][2] = {};
    for (int k0 = 0; k0 < 768; k0 += 32) {
#pragma unroll
        for (int i = 0; i < 8; ++i) {                    // A tile 64x32 (implicit im2col gather)
            int e = tid + i * 256;
            int r = e >> 5, kk = e & 31;
            int m = bm * 64 + r;
            int k = k0 + kk;
            int ky = k / 48; int rem = k - ky * 48; int kx = rem / 3; int c = rem - kx * 3;
            int oy = m >> 6, ox = m & 63;
            As_f[a_slot(r, kk)] =
                img[((size_t)(oy * 16 + ky) * 1024 + (ox * 16 + kx)) * 3 + c];
        }
        __syncthreads();

        v16h afrag[2], bfrag[2];
#pragma unroll
        for (int ms = 0; ms < 2; ++ms)
            afrag[ms] = load_frag(&As_f[((wm * 2 + ms) * 32 + lane) * FRAG_STRIDE]);
#pragma unroll
        for (int ns = 0; ns < 2; ++ns)
            bfrag[ns] = load_frag(
                &bpack[(((size_t)(k0 >> 5) * 32 + bn * 8 + wn * 2 + ns) * 32 + lane) * 16]);
#pragma unroll
        for (int ms = 0; ms < 2; ++ms)
#pragma unroll
            for (int ns = 0; ns < 2; ++ns)
                acc[ms][ns] = __builtin_amdgcn_wmma_f32_16x16x32_f16(
                    false, afrag[ms], false, bfrag[ns], (short)0, acc[ms][ns], false, false);
        __syncthreads();
    }
#pragma unroll
    for (int ms = 0; ms < 2; ++ms)
#pragma unroll
        for (int ns = 0; ns < 2; ++ns) {
            int col = bn * 128 + (wn * 2 + ns) * 16 + (lane & 15);
            int rowb = bm * 64 + (wm * 2 + ms) * 16 + ((lane >> 4) << 3);
            float bv = bias[col];
#pragma unroll
            for (int r = 0; r < 8; ++r) {
                float v = acc[ms][ns][r] + bv;
                v = v > 0.f ? v : 0.f;
                out[(size_t)(rowb + r) * 512 + col] = (_Float16)v;
            }
        }
}

// rpn 3x3 conv as GEMM: M=3844, K=4608, N=512. A from feat f16 [64,64,512]; a 32-wide K slice
// never crosses a (ky,kx) block, so each thread stages A with one b128 load + one b128 ds store.
__global__ __launch_bounds__(256) void gemm_rpn(
    const _Float16* __restrict__ feat, const _Float16* __restrict__ bpack,
    const float* __restrict__ bias, _Float16* __restrict__ out)
{
    __shared__ _Float16 As_f[AS_HALVES];
    const int tid = threadIdx.x;
    const int bm = blockIdx.x, bn = blockIdx.y;
    const int wave = tid >> 5, lane = tid & 31;
    const int wm = wave & 1, wn = wave >> 1;

    const int ar = tid >> 2;                 // row this thread stages
    const int akk0 = (tid & 3) * 8;          // K-octet within the 32-slice
    const int am = bm * 64 + ar;
    const int aoy = am / FW, aox = am - aoy * FW;
    const int amsub = ar >> 4, amm = ar & 15, ag = akk0 >> 3;
    _Float16* adst = &As_f[((amsub * 32) + amm + (ag & 1) * 16) * FRAG_STRIDE + (ag >> 1) * 8];

    v8f acc[2][2] = {};
    for (int k0 = 0; k0 < 4608; k0 += 32) {
        {
            v8h av = {};
            if (am < NPOS) {
                int k = k0 + akk0;
                int ky = k / 1536; int rem = k - ky * 1536; int kx = rem >> 9; int ci = rem & 511;
                av = *(const v8h*)&feat[((size_t)(aoy + ky) * 64 + (aox + kx)) * 512 + ci];
            }
            *(v8h*)adst = av;
        }
        __syncthreads();

        v16h afrag[2], bfrag[2];
#pragma unroll
        for (int ms = 0; ms < 2; ++ms)
            afrag[ms] = load_frag(&As_f[((wm * 2 + ms) * 32 + lane) * FRAG_STRIDE]);
#pragma unroll
        for (int ns = 0; ns < 2; ++ns)
            bfrag[ns] = load_frag(
                &bpack[(((size_t)(k0 >> 5) * 32 + bn * 8 + wn * 2 + ns) * 32 + lane) * 16]);
#pragma unroll
        for (int ms = 0; ms < 2; ++ms)
#pragma unroll
            for (int ns = 0; ns < 2; ++ns)
                acc[ms][ns] = __builtin_amdgcn_wmma_f32_16x16x32_f16(
                    false, afrag[ms], false, bfrag[ns], (short)0, acc[ms][ns], false, false);
        __syncthreads();
    }
#pragma unroll
    for (int ms = 0; ms < 2; ++ms)
#pragma unroll
        for (int ns = 0; ns < 2; ++ns) {
            int col = bn * 128 + (wn * 2 + ns) * 16 + (lane & 15);
            int rowb = bm * 64 + (wm * 2 + ms) * 16 + ((lane >> 4) << 3);
            float bv = bias[col];
#pragma unroll
            for (int r = 0; r < 8; ++r) {
                int row = rowb + r;
                if (row < NPOS)
                    out[(size_t)row * 512 + col] = (_Float16)(acc[ms][ns][r] + bv);
            }
        }
}

// ---------------------------------------------------------------- RPN heads (1x1 convs)
__global__ __launch_bounds__(64) void rpn_heads_kernel(
    const _Float16* __restrict__ rpn, const float* __restrict__ wc, const float* __restrict__ bc,
    const float* __restrict__ wr, const float* __restrict__ br,
    float* __restrict__ probs, float* __restrict__ deltas)
{
    __shared__ _Float16 fv[512];
    const int m = blockIdx.x, t = threadIdx.x;
#pragma unroll
    for (int i = 0; i < 8; ++i) fv[t + i * 64] = rpn[(size_t)m * 512 + t + i * 64];
    __syncthreads();
    if (t < 9) {
        float acc = 0.f;
        for (int ci = 0; ci < 512; ++ci) acc += (float)fv[ci] * wc[ci * 9 + t];
        float lg = acc + bc[t];
        probs[m * 9 + t] = 1.f / (1.f + expf(-lg));
    } else if (t < 45) {
        int q = t - 9;
        float acc = 0.f;
        for (int ci = 0; ci < 512; ++ci) acc += (float)fv[ci] * wr[ci * 36 + q];
        deltas[(size_t)m * 36 + q] = acc + br[q];
    }
}

// ---------------------------------------------------------------- apply_deltas helper
__device__ inline float clip01f(float v) { return fminf(fmaxf(v, 0.f), 1.f); }

__device__ inline void apply_deltas1(const float b[4], const float d[4], float o[4]) {
    float w = b[2] - b[0], h = b[3] - b[1];
    float cx = b[0] + 0.5f * w, cy = b[1] + 0.5f * h;
    float ncx = cx + d[0] * w, ncy = cy + d[1] * h;
    float dw = fminf(fmaxf(d[2], -4.f), 4.f), dh = fminf(fmaxf(d[3], -4.f), 4.f);
    float nw = w * expf(dw), nh = h * expf(dh);
    o[0] = clip01f(ncx - 0.5f * nw); o[1] = clip01f(ncy - 0.5f * nh);
    o[2] = clip01f(ncx + 0.5f * nw); o[3] = clip01f(ncy + 0.5f * nh);
}

// ---------------------------------------------------------------- anchors + deltas -> boxes, score init
__global__ void make_boxes_kernel(const float* __restrict__ probs, const float* __restrict__ deltas,
                                  float* __restrict__ boxes, float* __restrict__ sc)
{
    int i = blockIdx.x * blockDim.x + threadIdx.x;
    if (i >= NANCH) return;
    int m = i / 9, a = i - m * 9;
    int y = m / FW, x = m - y * FW;
    const float SCALES[3] = {64.f, 128.f, 256.f};
    const float SQR[3] = {0.70710678f, 1.f, 1.41421356f};
    int s = a % 3, r = a / 3;
    float sx = 1024.f / FW, sy = 1024.f / FH;
    float cx = (x + 0.5f) * sx, cy = (y + 0.5f) * sy;
    float aw = SCALES[s] * SQR[r], ah = SCALES[s] / SQR[r];
    float b[4] = {(cx - 0.5f * aw) / 1024.f, (cy - 0.5f * ah) / 1024.f,
                  (cx + 0.5f * aw) / 1024.f, (cy + 0.5f * ah) / 1024.f};
    float d[4] = {deltas[(size_t)i * 4 + 0], deltas[(size_t)i * 4 + 1],
                  deltas[(size_t)i * 4 + 2], deltas[(size_t)i * 4 + 3]};
    float o[4];
    apply_deltas1(b, d, o);
    boxes[(size_t)i * 4 + 0] = o[0]; boxes[(size_t)i * 4 + 1] = o[1];
    boxes[(size_t)i * 4 + 2] = o[2]; boxes[(size_t)i * 4 + 3] = o[3];
    float p = probs[i];
    sc[i] = (p >= 0.4f) ? p : -1.f;
}

// ---------------------------------------------------------------- sequential NMS (matches lax.scan)
__global__ __launch_bounds__(256) void nms_kernel(
    const float* __restrict__ boxes, float* __restrict__ sc, const int* __restrict__ labels,
    float* __restrict__ out_b, float* __restrict__ out_s, float* __restrict__ out_l,
    int n, int top_n, float iou_th)
{
    __shared__ float sV[256];
    __shared__ int   sI[256];
    __shared__ float cb[4];
    __shared__ float cv;
    __shared__ int   cidx;
    const int t = threadIdx.x;

    for (int it = 0; it < top_n; ++it) {
        float bv = -3.0e38f; int bi = n;
        for (int j = t; j < n; j += 256) {
            float v = sc[j];
            if (v > bv || (v == bv && j < bi)) { bv = v; bi = j; }
        }
        sV[t] = bv; sI[t] = bi;
        __syncthreads();
        for (int stp = 128; stp > 0; stp >>= 1) {
            if (t < stp) {
                if (sV[t + stp] > sV[t] || (sV[t + stp] == sV[t] && sI[t + stp] < sI[t])) {
                    sV[t] = sV[t + stp]; sI[t] = sI[t + stp];
                }
            }
            __syncthreads();
        }
        if (t == 0) {
            cidx = sI[0]; cv = sV[0];
            cb[0] = boxes[(size_t)cidx * 4 + 0]; cb[1] = boxes[(size_t)cidx * 4 + 1];
            cb[2] = boxes[(size_t)cidx * 4 + 2]; cb[3] = boxes[(size_t)cidx * 4 + 3];
        }
        __syncthreads();
        float b0 = cb[0], b1 = cb[1], b2 = cb[2], b3 = cb[3];
        float a1 = fmaxf(b2 - b0, 0.f) * fmaxf(b3 - b1, 0.f);
        int ci = cidx;
        for (int j = t; j < n; j += 256) {
            float x1 = fmaxf(b0, boxes[(size_t)j * 4 + 0]);
            float y1 = fmaxf(b1, boxes[(size_t)j * 4 + 1]);
            float x2 = fminf(b2, boxes[(size_t)j * 4 + 2]);
            float y2 = fminf(b3, boxes[(size_t)j * 4 + 3]);
            float inter = fmaxf(x2 - x1, 0.f) * fmaxf(y2 - y1, 0.f);
            float a2 = fmaxf(boxes[(size_t)j * 4 + 2] - boxes[(size_t)j * 4 + 0], 0.f) *
                       fmaxf(boxes[(size_t)j * 4 + 3] - boxes[(size_t)j * 4 + 1], 0.f);
            float iou = inter / (a1 + a2 - inter + 1e-8f);
            if (iou > iou_th || j == ci) sc[j] = -1.f;
        }
        if (t == 0) {
            bool valid = cv > 0.f;
            out_b[it * 4 + 0] = valid ? b0 : 0.f;
            out_b[it * 4 + 1] = valid ? b1 : 0.f;
            out_b[it * 4 + 2] = valid ? b2 : 0.f;
            out_b[it * 4 + 3] = valid ? b3 : 0.f;
            out_s[it] = valid ? cv : 0.f;
            if (out_l) out_l[it] = valid ? (float)labels[ci] : -1.f;
        }
        __syncthreads();
    }
}

// ---------------------------------------------------------------- PS maps (3x3 SAME convs, 27 + 36 out ch)
__global__ __launch_bounds__(64) void psmap_kernel(
    const _Float16* __restrict__ feat,
    const float* __restrict__ wfc, const float* __restrict__ bfc,
    const float* __restrict__ wfr, const float* __restrict__ bfr,
    float* __restrict__ cmap, float* __restrict__ rmap)
{
    __shared__ _Float16 patch[9 * 512];
    const int m = blockIdx.x, t = threadIdx.x;
    const int oy = m >> 6, ox = m & 63;
#pragma unroll
    for (int i = 0; i < 72; ++i) {
        int e = t + i * 64;
        int kk = e >> 9, ci = e & 511;
        int ky = kk / 3, kx = kk - ky * 3;
        int y = oy + ky - 1, x = ox + kx - 1;
        _Float16 v = (_Float16)0.f;
        if (y >= 0 && y < 64 && x >= 0 && x < 64)
            v = feat[((size_t)y * 64 + x) * 512 + ci];
        patch[e] = v;
    }
    __syncthreads();
    if (t < 27) {
        float acc = 0.f;
        for (int e = 0; e < 4608; ++e) acc += (float)patch[e] * wfc[(size_t)e * 27 + t];
        cmap[(size_t)m * 27 + t] = acc + bfc[t];
    } else if (t < 63) {
        int co = t - 27;
        float acc = 0.f;
        for (int e = 0; e < 4608; ++e) acc += (float)patch[e] * wfr[(size_t)e * 36 + co];
        rmap[(size_t)m * 36 + co] = acc + bfr[co];
    }
}

// ---------------------------------------------------------------- PS-ROI align pooling
__global__ __launch_bounds__(128) void psroi_kernel(
    const float* __restrict__ cmap, const float* __restrict__ rmap,
    const float* __restrict__ rois, float* __restrict__ csc, float* __restrict__ bbp)
{
    __shared__ float accC[3];
    __shared__ float accR[4];
    const int r = blockIdx.x, t = threadIdx.x;
    if (t < 3) accC[t] = 0.f;
    if (t < 4) accR[t] = 0.f;
    __syncthreads();
    float x1 = rois[r * 4 + 0], y1 = rois[r * 4 + 1];
    float x2 = rois[r * 4 + 2], y2 = rois[r * 4 + 3];
    for (int s = t; s < 900; s += 128) {
        int p = s / 100, sm = s - p * 100;
        int py = p / 3, px = p - py * 3;
        int gy = py * 10 + sm / 10, gx = px * 10 + sm % 10;
        float xs = (x1 + (gx + 0.5f) / 30.f * (x2 - x1)) * 63.f;
        float ys = (y1 + (gy + 0.5f) / 30.f * (y2 - y1)) * 63.f;
        float fx0 = fminf(fmaxf(floorf(xs), 0.f), 62.f);
        float fy0 = fminf(fmaxf(floorf(ys), 0.f), 62.f);
        int ix = (int)fx0, iy = (int)fy0;
        float fx = fminf(fmaxf(xs - fx0, 0.f), 1.f);
        float fy = fminf(fmaxf(ys - fy0, 0.f), 1.f);
        float w00 = (1.f - fy) * (1.f - fx), w01 = (1.f - fy) * fx;
        float w10 = fy * (1.f - fx),        w11 = fy * fx;
        size_t i00c = ((size_t)iy * 64 + ix) * 27, i01c = i00c + 27;
        size_t i10c = i00c + 64 * 27,             i11c = i10c + 27;
#pragma unroll
        for (int j = 0; j < 3; ++j) {
            int ch = j * 9 + p;
            float v = cmap[i00c + ch] * w00 + cmap[i01c + ch] * w01 +
                      cmap[i10c + ch] * w10 + cmap[i11c + ch] * w11;
            atomicAdd(&accC[j], v);
        }
        size_t i00r = ((size_t)iy * 64 + ix) * 36, i01r = i00r + 36;
        size_t i10r = i00r + 64 * 36,             i11r = i10r + 36;
#pragma unroll
        for (int j = 0; j < 4; ++j) {
            int ch = j * 9 + p;
            float v = rmap[i00r + ch] * w00 + rmap[i01r + ch] * w01 +
                      rmap[i10r + ch] * w10 + rmap[i11r + ch] * w11;
            atomicAdd(&accR[j], v);
        }
    }
    __syncthreads();
    if (t < 3) csc[r * 3 + t] = accC[t] / 900.f;
    if (t < 4) bbp[r * 4 + t] = accR[t] / 900.f;
}

// ---------------------------------------------------------------- softmax / score / deltas for 2nd stage
__global__ __launch_bounds__(128) void finalize_kernel(
    const float* __restrict__ csc, const float* __restrict__ bbp, const float* __restrict__ rois,
    float* __restrict__ fboxes, float* __restrict__ fsc, int* __restrict__ flab)
{
    int r = threadIdx.x;
    if (r >= 100) return;
    float s0 = csc[r * 3 + 0], s1 = csc[r * 3 + 1], s2 = csc[r * 3 + 2];
    float mx = fmaxf(s0, fmaxf(s1, s2));
    float e0 = expf(s0 - mx), e1 = expf(s1 - mx), e2 = expf(s2 - mx);
    float inv = 1.f / (e0 + e1 + e2);
    float p0 = e0 * inv, p1 = e1 * inv;
    float score = fmaxf(p0, p1);
    int label = (p1 > p0) ? 1 : 0;
    float b[4] = {rois[r * 4 + 0], rois[r * 4 + 1], rois[r * 4 + 2], rois[r * 4 + 3]};
    float d[4] = {bbp[r * 4 + 0], bbp[r * 4 + 1], bbp[r * 4 + 2], bbp[r * 4 + 3]};
    float o[4];
    apply_deltas1(b, d, o);
    fboxes[r * 4 + 0] = o[0]; fboxes[r * 4 + 1] = o[1];
    fboxes[r * 4 + 2] = o[2]; fboxes[r * 4 + 3] = o[3];
    fsc[r] = (score >= 0.5f) ? score : -1.f;
    flab[r] = label;
}

// ---------------------------------------------------------------- launch
extern "C" void kernel_launch(void* const* d_in, const int* in_sizes, int n_in,
                              void* d_out, int out_size, void* d_ws, size_t ws_size,
                              hipStream_t stream) {
    (void)in_sizes; (void)n_in; (void)out_size; (void)ws_size;
    const float* image = (const float*)d_in[0];
    const float* wb  = (const float*)d_in[1];
    const float* bb  = (const float*)d_in[2];
    const float* wi  = (const float*)d_in[3];
    const float* bi  = (const float*)d_in[4];
    const float* wc  = (const float*)d_in[5];
    const float* bc  = (const float*)d_in[6];
    const float* wr  = (const float*)d_in[7];
    const float* br  = (const float*)d_in[8];
    const float* wfc = (const float*)d_in[9];
    const float* bfc = (const float*)d_in[10];
    const float* wfr = (const float*)d_in[11];
    const float* bfr = (const float*)d_in[12];

    char* ws = (char*)d_ws;
    _Float16* img_h  = (_Float16*)(ws + O_IMGH);
    _Float16* w1_p   = (_Float16*)(ws + O_W1P);
    _Float16* feat_h = (_Float16*)(ws + O_FEATH);
    _Float16* w2_p   = (_Float16*)(ws + O_W2P);
    _Float16* rpn_h  = (_Float16*)(ws + O_RPNH);
    float* probs  = (float*)(ws + O_PROB);
    float* deltas = (float*)(ws + O_DELT);
    float* boxes  = (float*)(ws + O_BOX);
    float* sc     = (float*)(ws + O_SC);
    float* cmap   = (float*)(ws + O_CMAP);
    float* rmap   = (float*)(ws + O_RMAP);
    float* csc    = (float*)(ws + O_CSC);
    float* bbp    = (float*)(ws + O_BBP);
    float* fboxes = (float*)(ws + O_FBX);
    float* fsc    = (float*)(ws + O_FSC);
    int*   flab   = (int*)(ws + O_FLB);

    float* out  = (float*)d_out;
    float* rois = out;            // [100,4]
    float* rpnp = out + 400;      // [100]
    float* outb = out + 500;      // [10,4]
    float* outs = out + 540;      // [10]
    float* outl = out + 550;      // [10]

    cvt_f16_kernel<<<(int)((IMG_N + 255) / 256), 256, 0, stream>>>(image, img_h, (int)IMG_N);
    pack_w_kernel<<<(int)((W1_N + 255) / 256), 256, 0, stream>>>(wb, w1_p, (int)W1_N);
    pack_w_kernel<<<(int)((W2_N + 255) / 256), 256, 0, stream>>>(wi, w2_p, (int)W2_N);

    gemm_backbone<<<dim3(64, 4), 256, 0, stream>>>(img_h, w1_p, bb, feat_h);
    gemm_rpn<<<dim3(61, 4), 256, 0, stream>>>(feat_h, w2_p, bi, rpn_h);

    rpn_heads_kernel<<<NPOS, 64, 0, stream>>>(rpn_h, wc, bc, wr, br, probs, deltas);
    make_boxes_kernel<<<(NANCH + 255) / 256, 256, 0, stream>>>(probs, deltas, boxes, sc);
    nms_kernel<<<1, 256, 0, stream>>>(boxes, sc, nullptr, rois, rpnp, nullptr,
                                      NANCH, 100, 0.5f);

    psmap_kernel<<<4096, 64, 0, stream>>>(feat_h, wfc, bfc, wfr, bfr, cmap, rmap);
    psroi_kernel<<<100, 128, 0, stream>>>(cmap, rmap, rois, csc, bbp);
    finalize_kernel<<<1, 128, 0, stream>>>(csc, bbp, rois, fboxes, fsc, flab);
    nms_kernel<<<1, 256, 0, stream>>>(fboxes, fsc, flab, outb, outs, outl,
                                      100, 10, 0.1f);
}
